// MultiHeadAttention_24129126269141
// MI455X (gfx1250) — matrix operand order repaired and checked by reading, not compiled, not run on hardware
//
#include <hip/hip_runtime.h>
#include <hip/hip_bf16.h>
#include <math.h>

// Problem constants (from reference): B=2, S=1024, D=1024, H=16, DH=64
#define B_  2
#define S_  1024
#define D_  1024
#define H_  16
#define DH_ 64
#define M_  (B_*S_)       // 2048 rows for all GEMMs
#define MAXD_ 31
#define NEGV (-1e9f)

typedef unsigned short u16;
typedef unsigned int   u32;
typedef unsigned long long u64;
typedef __attribute__((ext_vector_type(16))) __bf16 v16bf;
typedef __attribute__((ext_vector_type(8)))  float  v8f;

union FragB16 { u32 u[8]; v16bf v; };

// CDNA5 async global->LDS staging (ASYNCcnt-tracked, §10 VGLOBAL async ops).
// IOFFSET is added to BOTH the LDS and global addresses (ISA 08 §4.4), so we
// fold per-row strides into the per-lane base addresses and keep offsets
// contiguous. All transfers are 16B aligned on both sides.
#define ASYNC_B128(ldsaddr, gaddr, OFFS)                                      \
  asm volatile("global_load_async_to_lds_b128 %0, %1, off offset:" OFFS      \
               :: "v"(ldsaddr), "v"(gaddr) : "memory")
#define WAIT_ASYNC() asm volatile("s_wait_asynccnt 0" ::: "memory")

__device__ __forceinline__ u32 lds_addr(const void* p) {
  return (u32)(unsigned long long)p;   // low 32 bits of flat addr == LDS offset
}

__device__ __forceinline__ u16 f2bf(float f) {
  union { float f; u32 u; } x; x.f = f;
  u32 r = x.u + 0x7FFFu + ((x.u >> 16) & 1u);   // round-to-nearest-even
  return (u16)(r >> 16);
}
__device__ __forceinline__ float bf2f(u16 x) {
  union { u32 u; float f; } t; t.u = ((u32)x) << 16; return t.f;
}
__device__ __forceinline__ v8f zero8() {
  v8f z;
  for (int i = 0; i < 8; ++i) z[i] = 0.f;
  return z;
}

// ---------------------------------------------------------------------------
// fp32 -> bf16 cast (activations, A-side operands)
// ---------------------------------------------------------------------------
__global__ void cast_f32_to_bf16(const float* __restrict__ src, u16* __restrict__ dst, long n) {
  long i = (long)blockIdx.x * blockDim.x + threadIdx.x;
  long stride = (long)gridDim.x * blockDim.x;
  for (; i < n; i += stride) dst[i] = f2bf(src[i]);
}

// ---------------------------------------------------------------------------
// Weight pre-pack: W (K x N, fp32, row-major) -> Wp (N x K/2, u32), where
// Wp[n][kk] = bf16(W[2kk][n]) | bf16(W[2kk+1][n]) << 16.
// This is exactly the per-VGPR K-pair layout of the WMMA 16-bit B operand, so
// B fragments become 8 contiguous dwords (2x ds_load_b128), no packing VALU.
// ---------------------------------------------------------------------------
__global__ void pack_weight(const float* __restrict__ W, u32* __restrict__ Wp,
                            int K, int N) {
  const long n = (long)N * (K >> 1);
  long i = (long)blockIdx.x * blockDim.x + threadIdx.x;
  const long stride = (long)gridDim.x * blockDim.x;
  for (; i < n; i += stride) {
    const int kk = (int)(i % (K >> 1));
    const int nn = (int)(i / (K >> 1));
    const u32 lo = f2bf(W[(size_t)(2 * kk) * N + nn]);
    const u32 hi = f2bf(W[(size_t)(2 * kk + 1) * N + nn]);
    Wp[i] = lo | (hi << 16);
  }
}

// ---------------------------------------------------------------------------
// V pre-pack for attention: Vh ([B,H,S,DH] bf16) -> Vp ([B,H,DH,S/2] u32),
// key-pair packed; optional key permutation (column view) fused in.
// ---------------------------------------------------------------------------
__global__ void pack_v_bf16(const u16* __restrict__ Vh, const int* __restrict__ perm,
                            u32* __restrict__ Vp) {
  const size_t n = (size_t)B_ * H_ * DH_ * (S_ / 2);
  size_t i = (size_t)blockIdx.x * blockDim.x + threadIdx.x;
  const size_t stride = (size_t)gridDim.x * blockDim.x;
  for (; i < n; i += stride) {
    const int kk  = (int)(i & (S_ / 2 - 1));       // 9 bits
    const int dim = (int)((i >> 9) & (DH_ - 1));   // 6 bits
    const int bh  = (int)(i >> 15);
    const int b   = bh >> 4;
    int k0 = 2 * kk, k1 = 2 * kk + 1;
    if (perm) { k0 = perm[b * S_ + k0]; k1 = perm[b * S_ + k1]; }
    const u32 lo = Vh[((size_t)bh * S_ + k0) * DH_ + dim];
    const u32 hi = Vh[((size_t)bh * S_ + k1) * DH_ + dim];
    Vp[i] = lo | (hi << 16);
  }
}

// ---------------------------------------------------------------------------
// bf16 WMMA GEMM: C = A(MxK) * Bp + bias, fp32 accumulation.
// A: bf16 row-major; Bp: pre-packed [N][K/2] u32 (see pack_weight).
// Block tile 128x128 (8 waves, each 32x64 => 2x4 wmma accumulators).
// Tiles staged via global_load_async_to_lds_b128 (ASYNCcnt).
// Requires: M % 128 == 0, K % 32 == 0. N arbitrary (guarded).
// ---------------------------------------------------------------------------
#define GEMM_OUT_F32   0
#define GEMM_OUT_BF16  1
#define GEMM_OUT_HEADS 2   // scatter (b*S+s, h*DH+dh) -> [B,H,S,DH] bf16

#define BLK_M 128
#define BLK_N 128
#define BLK_K 32
#define STRA  40    // LDS row stride (u16) for A tile (128x32); 80B, 16B-aligned
#define STRBP 20    // LDS row stride (u32) for packed B tile (128n x 16kk); 80B

__launch_bounds__(256)
__global__ void gemm_bf16_wmma(const u16* __restrict__ A, const u32* __restrict__ Bp,
                               const float* __restrict__ bias, void* __restrict__ out,
                               int Mdim, int Ndim, int Kdim, int mode) {
  __shared__ u16 As[BLK_M * STRA];
  __shared__ u32 Bcm[BLK_N * STRBP];

  const int t    = threadIdx.x;
  const int lane = t & 31;
  const int wid  = t >> 5;
  const int mBase = blockIdx.y * BLK_M;
  const int nBase = blockIdx.x * BLK_N;
  const int mOff  = (wid & 3) * 32;
  const int nOff  = (wid >> 2) * 64;

  v8f acc[2][4];
  for (int mi = 0; mi < 2; ++mi)
    for (int ni = 0; ni < 4; ++ni) acc[mi][ni] = zero8();

  const int aRow = t >> 1, aCol = (t & 1) * 16;   // A: 128 rows x 32 cols (u16)
  const int bN   = t >> 1, bHalf = (t & 1) * 8;   // B: 128 n-rows x 16 kk (u32)
  const int gN   = nBase + bN;
  const int Khalf = Kdim >> 1;

  for (int k0 = 0; k0 < Kdim; k0 += BLK_K) {
    // --- stage A tile (async, 32B per thread) ---
    {
      const u16* gp = A + (size_t)(mBase + aRow) * Kdim + k0 + aCol;
      const u32 dl = lds_addr(&As[aRow * STRA + aCol]);
      const u64 ga = (u64)gp;
      ASYNC_B128(dl, ga, "0");
      ASYNC_B128(dl, ga, "16");
      if (k0 + BLK_K < Kdim) __builtin_prefetch(gp + BLK_K, 0, 1);
    }
    // --- stage packed B tile (async, 32B per thread; zero-fill OOB n) ---
    {
      u32* dp = &Bcm[bN * STRBP + bHalf];
      if (gN < Ndim) {
        const u32* gp = Bp + (size_t)gN * Khalf + (k0 >> 1) + bHalf;
        const u32 dl = lds_addr(dp);
        const u64 ga = (u64)gp;
        ASYNC_B128(dl, ga, "0");
        ASYNC_B128(dl, ga, "16");
        if (k0 + BLK_K < Kdim) __builtin_prefetch(gp + 16, 0, 1);
      } else {
        for (int i = 0; i < 8; ++i) dp[i] = 0u;
      }
    }
    WAIT_ASYNC();
    __syncthreads();

    // --- A fragments (16-bit A 16x32, contiguous k: lane half h holds k 16h..16h+15) ---
    FragB16 af[2];
    {
      const int r0 = mOff + (lane & 15);
      const int kb = (lane < 16) ? 0 : 16;
      for (int mi = 0; mi < 2; ++mi) {
        const u16* base = &As[(r0 + mi * 16) * STRA];
        for (int j = 0; j < 4; ++j) af[mi].u[j]     = *(const u32*)&base[kb + 2 * j];
        for (int j = 0; j < 4; ++j) af[mi].u[4 + j] = *(const u32*)&base[kb + 8 + 2 * j];
      }
    }
    // --- B fragments: 8 contiguous dwords each (2x ds_load_b128) ---
    FragB16 bf[4];
    {
      const int n0 = nOff + (lane & 15);
      const int kp = (lane < 16) ? 0 : 8;   // packed k-pair offset
      for (int ni = 0; ni < 4; ++ni) {
        const u32* bb = &Bcm[(n0 + ni * 16) * STRBP + kp];
        for (int j = 0; j < 8; ++j) bf[ni].u[j] = bb[j];
      }
    }
    for (int mi = 0; mi < 2; ++mi)
      for (int ni = 0; ni < 4; ++ni)
        acc[mi][ni] = __builtin_amdgcn_wmma_f32_16x16x32_bf16(
            false, af[mi].v, false, bf[ni].v, (short)0, acc[mi][ni], false, false);
    __syncthreads();
  }

  // --- epilogue: C layout VGPR r -> rows r / r+8, lanes split columns ---
  for (int mi = 0; mi < 2; ++mi)
    for (int ni = 0; ni < 4; ++ni)
      for (int r = 0; r < 8; ++r) {
        const int row = mBase + mOff + mi * 16 + r + ((lane < 16) ? 0 : 8);
        const int col = nBase + nOff + ni * 16 + (lane & 15);
        if (row >= Mdim || col >= Ndim) continue;
        float vv = acc[mi][ni][r] + (bias ? bias[col] : 0.f);
        if (mode == GEMM_OUT_F32) {
          ((float*)out)[(size_t)row * Ndim + col] = vv;
        } else if (mode == GEMM_OUT_BF16) {
          ((u16*)out)[(size_t)row * Ndim + col] = f2bf(vv);
        } else {  // GEMM_OUT_HEADS
          const int bb = row >> 10, s = row & (S_ - 1);
          const int h  = col >> 6,  dh = col & (DH_ - 1);
          ((u16*)out)[(((size_t)bb * H_ + h) * S_ + s) * DH_ + dh] = f2bf(vv);
        }
      }
}

// ---------------------------------------------------------------------------
// Flash attention, one wave per 16-query tile, 32-key blocks.
// Q/K staged async; V consumed from pre-packed Vp ([B,H,DH,S/2] u32).
// mode 0: dense bias posBias[h,q,k]; mode 1: rel_table[clip(pc[k]-pc[q])][h].
// Causal + key-padding handled as value replacement with -1e9 (as reference).
// ---------------------------------------------------------------------------
#define FQ_STR 72   // u16 row stride for Q/K tiles (144B, 16B-aligned)
#define FV_STR 20   // u32 row stride for packed V tile (80B, 16B-aligned)
#define FP_STR 40   // u16 row stride for P tile

__launch_bounds__(32)
__global__ void flash_attn_wmma(const u16* __restrict__ Qh, const u16* __restrict__ Kh,
                                const u32* __restrict__ Vp, float* __restrict__ ctxOut,
                                const float* __restrict__ posBias,   // mode 0
                                const float* __restrict__ relTable,  // mode 1
                                const int*   __restrict__ pc,        // mode 1
                                const float* __restrict__ padKey,    // [B*S]
                                int mode) {
  __shared__ u16 qs[16 * FQ_STR];
  __shared__ u16 ks[32 * FQ_STR];
  __shared__ u32 vpk[64 * FV_STR];
  __shared__ u16 ps[16 * FP_STR];

  const int lane  = threadIdx.x & 31;
  const int qTile = blockIdx.x;
  const int h     = blockIdx.y;
  const int b     = blockIdx.z;
  const int qBase = qTile * 16;
  const int bh    = b * H_ + h;
  const size_t headOff = (size_t)bh * S_ * DH_;

  // stage Q tile (16 x 64 bf16), 64B per lane
  {
    const int row = lane >> 1, half = (lane & 1) * 32;
    const u64 ga = (u64)(Qh + headOff + (size_t)(qBase + row) * DH_ + half);
    const u32 dl = lds_addr(&qs[row * FQ_STR + half]);
    ASYNC_B128(dl, ga, "0");  ASYNC_B128(dl, ga, "16");
    ASYNC_B128(dl, ga, "32"); ASYNC_B128(dl, ga, "48");
  }
  WAIT_ASYNC();
  __syncthreads();

  FragB16 aq[2];
  {
    const u16* base = &qs[(lane & 15) * FQ_STR];
    const int kb = (lane < 16) ? 0 : 16;
    for (int d = 0; d < 2; ++d) {
      for (int j = 0; j < 4; ++j) aq[d].u[j]     = *(const u32*)&base[d * 32 + kb + 2 * j];
      for (int j = 0; j < 4; ++j) aq[d].u[4 + j] = *(const u32*)&base[d * 32 + kb + 8 + 2 * j];
    }
  }

  float mrow[8], lrow[8];
  v8f o[4];
  for (int r = 0; r < 8; ++r) { mrow[r] = -3.0e38f; lrow[r] = 0.f; }
  for (int ni = 0; ni < 4; ++ni) o[ni] = zero8();

  const float scale   = 0.125f;  // 1/sqrt(64)
  const int   rowHalf = (lane < 16) ? 0 : 8;
  const int   kbMax   = (qBase + 15) >> 5;

  for (int kb = 0; kb <= kbMax; ++kb) {
    // stage K block (32 x 64 bf16): one 128B row per lane, async
    {
      const u64 ga = (u64)(Kh + headOff + (size_t)(kb * 32 + lane) * DH_);
      const u32 dl = lds_addr(&ks[lane * FQ_STR]);
      ASYNC_B128(dl, ga, "0");  ASYNC_B128(dl, ga, "16");
      ASYNC_B128(dl, ga, "32"); ASYNC_B128(dl, ga, "48");
      ASYNC_B128(dl, ga, "64"); ASYNC_B128(dl, ga, "80");
      ASYNC_B128(dl, ga, "96"); ASYNC_B128(dl, ga, "112");
    }
    // stage packed V block (64 dims x 16 key-pairs u32): two 64B rows per lane
    {
      for (int rr = lane; rr < 64; rr += 32) {
        const u64 ga = (u64)(Vp + ((size_t)bh * DH_ + rr) * (S_ / 2) + kb * 16);
        const u32 dl = lds_addr(&vpk[rr * FV_STR]);
        ASYNC_B128(dl, ga, "0");  ASYNC_B128(dl, ga, "16");
        ASYNC_B128(dl, ga, "32"); ASYNC_B128(dl, ga, "48");
      }
    }
    WAIT_ASYNC();
    __syncthreads();

    // scores: S = Q * K^T  (two 16-key subtiles, 2 wmma each over DH=64)
    float sc2[2][8];
    for (int sn = 0; sn < 2; ++sn) {
      FragB16 bk[2];
      const int key = sn * 16 + (lane & 15);
      const int kbb = (lane < 16) ? 0 : 16;
      for (int d = 0; d < 2; ++d)
        for (int j = 0; j < 8; ++j)
          bk[d].u[j] = *(const u32*)&ks[(size_t)key * FQ_STR + d * 32 + kbb + 2 * j];

      v8f s = zero8();
      s = __builtin_amdgcn_wmma_f32_16x16x32_bf16(false, aq[0].v, false, bk[0].v, (short)0, s, false, false);
      s = __builtin_amdgcn_wmma_f32_16x16x32_bf16(false, aq[1].v, false, bk[1].v, (short)0, s, false, false);

      const int keyG = kb * 32 + sn * 16 + (lane & 15);
      for (int r = 0; r < 8; ++r) {
        const int qRow = qBase + r + rowHalf;
        float val = s[r] * scale;
        if (mode == 0) {
          val += posBias[((size_t)h * S_ + qRow) * S_ + keyG];
        } else {
          int dlt = pc[(size_t)b * S_ + keyG] - pc[(size_t)b * S_ + qRow];
          dlt = (dlt < -MAXD_) ? -MAXD_ : ((dlt > MAXD_) ? MAXD_ : dlt);
          val += relTable[(dlt + MAXD_) * H_ + h];
        }
        if (keyG > qRow || padKey[(size_t)b * S_ + keyG] == 0.f) val = NEGV;
        sc2[sn][r] = val;
      }
    }

    // online softmax per row (rows live across lanes 0..15 / 16..31)
    float pr[2][8];
    for (int r = 0; r < 8; ++r) {
      float mx = fmaxf(sc2[0][r], sc2[1][r]);
      for (int off = 1; off < 16; off <<= 1) mx = fmaxf(mx, __shfl_xor(mx, off, 16));
      const float mnew  = fmaxf(mrow[r], mx);
      const float alpha = __expf(mrow[r] - mnew);
      const float p0 = __expf(sc2[0][r] - mnew);
      const float p1 = __expf(sc2[1][r] - mnew);
      float rs = p0 + p1;
      for (int off = 1; off < 16; off <<= 1) rs += __shfl_xor(rs, off, 16);
      lrow[r] = lrow[r] * alpha + rs;
      mrow[r] = mnew;
      pr[0][r] = p0; pr[1][r] = p1;
      for (int ni = 0; ni < 4; ++ni) o[ni][r] *= alpha;
    }
    __syncthreads();
    for (int sn = 0; sn < 2; ++sn)
      for (int r = 0; r < 8; ++r)
        ps[(r + rowHalf) * FP_STR + sn * 16 + (lane & 15)] = f2bf(pr[sn][r]);
    __syncthreads();

    // O += P * V  (P: 16x32 A fragment; V: packed B fragments, contiguous)
    FragB16 ap;
    {
      const u16* base = &ps[(lane & 15) * FP_STR];
      const int kb2 = (lane < 16) ? 0 : 16;
      for (int j = 0; j < 4; ++j) ap.u[j]     = *(const u32*)&base[kb2 + 2 * j];
      for (int j = 0; j < 4; ++j) ap.u[4 + j] = *(const u32*)&base[kb2 + 8 + 2 * j];
    }
    const int kp = (lane < 16) ? 0 : 8;
    const int nd = lane & 15;
    for (int ni = 0; ni < 4; ++ni) {
      FragB16 bv;
      const u32* bb = &vpk[(ni * 16 + nd) * FV_STR + kp];
      for (int j = 0; j < 8; ++j) bv.u[j] = bb[j];
      o[ni] = __builtin_amdgcn_wmma_f32_16x16x32_bf16(false, ap.v, false, bv.v, (short)0, o[ni], false, false);
    }
    __syncthreads();  // protect ks/vpk before next block's async staging
  }

  for (int ni = 0; ni < 4; ++ni)
    for (int r = 0; r < 8; ++r) {
      const int row = qBase + r + rowHalf;
      ctxOut[headOff + (size_t)row * DH_ + ni * 16 + (lane & 15)] = o[ni][r] / lrow[r];
    }
}

// ---------------------------------------------------------------------------
// small helper kernels
// ---------------------------------------------------------------------------
__global__ void init_misc(float* entOut, int* maxv) {
  if (blockIdx.x == 0 && threadIdx.x == 0) {
    entOut[0] = 0.f;
    for (int b = 0; b < B_; ++b) maxv[b] = 0;
  }
}

__global__ void rowmax_kernel(const int* __restrict__ pos, int* __restrict__ maxv) {
  int i = blockIdx.x * blockDim.x + threadIdx.x;
  if (i < B_ * S_) atomicMax(&maxv[i / S_], pos[(size_t)i * 2]);
}

__global__ void perm_kernel(const int* __restrict__ pos, const int* __restrict__ maxv,
                            int* __restrict__ perm, int* __restrict__ invp) {
  int i = blockIdx.x * blockDim.x + threadIdx.x;
  if (i >= B_ * S_) return;
  const int b = i / S_, s = i % S_;
  const int row = pos[(size_t)i * 2], col = pos[(size_t)i * 2 + 1];
  const int comp = col * (maxv[b] + 1) + row;   // bijection -> argsort == scatter
  perm[b * S_ + comp] = s;
  invp[i] = comp;
}

__global__ void pcpad_kernel(const int* __restrict__ perm, const int* __restrict__ pos,
                             const float* __restrict__ pad, int* __restrict__ pc,
                             float* __restrict__ padc) {
  int i = blockIdx.x * blockDim.x + threadIdx.x;
  if (i >= B_ * S_) return;
  const int b = i / S_;
  const int p = perm[i];
  pc[i]   = pos[((size_t)b * S_ + p) * 2 + 1];
  padc[i] = pad[(size_t)b * S_ + p];
}

__global__ void gather_heads(const u16* __restrict__ src, const int* __restrict__ perm,
                             u16* __restrict__ dst) {
  const size_t n = (size_t)B_ * H_ * S_ * DH_;
  size_t i = (size_t)blockIdx.x * blockDim.x + threadIdx.x;
  const size_t stride = (size_t)gridDim.x * blockDim.x;
  for (; i < n; i += stride) {
    const size_t dh = i & (DH_ - 1);
    const size_t j  = (i >> 6) & (S_ - 1);
    const size_t h  = (i >> 16) & (H_ - 1);
    const size_t b  = i >> 20;
    const int srcS = perm[b * S_ + j];
    dst[i] = src[(((b * H_ + h) * S_) + srcS) * DH_ + dh];
  }
}

__global__ void silu_mul_kernel(const u16* __restrict__ up, const u16* __restrict__ gate,
                                u16* __restrict__ act, long n) {
  long i = (long)blockIdx.x * blockDim.x + threadIdx.x;
  const long stride = (long)gridDim.x * blockDim.x;
  for (; i < n; i += stride) {
    const float u = bf2f(up[i]);
    const float g = bf2f(gate[i]);
    const float s = u / (1.f + __expf(-u));
    act[i] = f2bf(s * g);
  }
}

__global__ void router_softmax_entropy(const float* __restrict__ logits, float* __restrict__ rw,
                                       float* __restrict__ entOut) {
  const int n = M_ * H_;
  int i = blockIdx.x * blockDim.x + threadIdx.x;
  if (i >= n) return;
  const float l0 = logits[2 * i], l1 = logits[2 * i + 1];
  const float mx = fmaxf(l0, l1);
  const float e0 = __expf(l0 - mx), e1 = __expf(l1 - mx);
  const float z = e0 + e1;
  const float r0 = e0 / z, r1 = e1 / z;
  rw[2 * i] = r0; rw[2 * i + 1] = r1;
  const float ent = -(r0 * __logf(r0 + 1e-9f) + r1 * __logf(r1 + 1e-9f));
  atomicAdd(entOut, ent / (float)n);
}

__global__ void combine_kernel(const float* __restrict__ ctxRow, const float* __restrict__ ctxColP,
                               const float* __restrict__ rw, const int* __restrict__ invp,
                               u16* __restrict__ ctxOut) {
  const size_t n = (size_t)B_ * S_ * D_;
  size_t i = (size_t)blockIdx.x * blockDim.x + threadIdx.x;
  const size_t stride = (size_t)gridDim.x * blockDim.x;
  for (; i < n; i += stride) {
    const size_t col = i & (D_ - 1);
    const size_t s   = (i >> 10) & (S_ - 1);
    const size_t b   = i >> 20;
    const int h = (int)(col >> 6), dh = (int)(col & (DH_ - 1));
    const float r0 = rw[(((b * S_ + s) * H_) + h) * 2];
    const float r1 = rw[(((b * S_ + s) * H_) + h) * 2 + 1];
    const int ip = invp[b * S_ + s];
    const float vr = ctxRow[(((b * H_ + h) * S_) + s) * DH_ + dh];
    const float vc = ctxColP[(((b * H_ + h) * S_) + (size_t)ip) * DH_ + dh];
    ctxOut[i] = f2bf(r0 * vr + r1 * vc);
  }
}

// ---------------------------------------------------------------------------
// host side
// ---------------------------------------------------------------------------
extern "C" void kernel_launch(void* const* d_in, const int* in_sizes, int n_in,
                              void* d_out, int out_size, void* d_ws, size_t ws_size,
                              hipStream_t stream) {
  (void)in_sizes; (void)n_in; (void)out_size; (void)ws_size;

  const float* query  = (const float*)d_in[0];
  const float* keyin  = (const float*)d_in[1];
  const float* value  = (const float*)d_in[2];
  const float* hidden = (const float*)d_in[3];
  const float* padM   = (const float*)d_in[4];   // [B,1,1,S]
  // d_in[5] causal mask: applied analytically
  const float* posBias = (const float*)d_in[6];  // [1,H,S,S]
  const int*   posIdx  = (const int*)d_in[7];    // [B,S,2]
  const float* Wq = (const float*)d_in[8];  const float* bq = (const float*)d_in[9];
  const float* Wk = (const float*)d_in[10]; const float* bk = (const float*)d_in[11];
  const float* Wv = (const float*)d_in[12]; const float* bv = (const float*)d_in[13];
  const float* Wo = (const float*)d_in[14]; const float* bo = (const float*)d_in[15];
  const float* Wup = (const float*)d_in[16]; const float* bup = (const float*)d_in[17];
  const float* Wgt = (const float*)d_in[18]; const float* bgt = (const float*)d_in[19];
  const float* Wdn = (const float*)d_in[20]; const float* bdn = (const float*)d_in[21];
  const float* relT = (const float*)d_in[22];

  float* outp = (float*)d_out;
  float* entOut = outp + (size_t)B_ * S_ * D_;

  // bump allocator over workspace
  char* wp = (char*)d_ws;
  auto alloc = [&](size_t bytes) -> void* {
    void* p = (void*)wp;
    wp += (bytes + 255) & ~(size_t)255;
    return p;
  };
  const size_t nBSD = (size_t)M_ * D_;
  u16* qbf  = (u16*)alloc(nBSD * 2);
  u16* kbf  = (u16*)alloc(nBSD * 2);
  u16* vbf  = (u16*)alloc(nBSD * 2);
  u16* hbf  = (u16*)alloc(nBSD * 2);
  u32* wqp  = (u32*)alloc((size_t)D_ * (D_ / 2) * 4);
  u32* wkp  = (u32*)alloc((size_t)D_ * (D_ / 2) * 4);
  u32* wvp  = (u32*)alloc((size_t)D_ * (D_ / 2) * 4);
  u32* wop  = (u32*)alloc((size_t)D_ * (D_ / 2) * 4);
  u32* wupp = (u32*)alloc((size_t)4 * D_ * (D_ / 2) * 4);
  u32* wgtp = (u32*)alloc((size_t)4 * D_ * (D_ / 2) * 4);
  u32* wdnp = (u32*)alloc((size_t)2 * H_ * (4 * D_ / 2) * 4);
  u16* Qh   = (u16*)alloc(nBSD * 2);
  u16* Kh   = (u16*)alloc(nBSD * 2);
  u16* Vh   = (u16*)alloc(nBSD * 2);
  u16* Qc   = (u16*)alloc(nBSD * 2);
  u16* Kc   = (u16*)alloc(nBSD * 2);
  u32* Vph  = (u32*)alloc(nBSD / 2 * 4);
  u32* Vpc  = (u32*)alloc(nBSD / 2 * 4);
  u16* upb  = (u16*)alloc((size_t)M_ * 4 * D_ * 2);
  u16* gtb  = (u16*)alloc((size_t)M_ * 4 * D_ * 2);
  u16* actb = (u16*)alloc((size_t)M_ * 4 * D_ * 2);
  float* logitsF = (float*)alloc((size_t)M_ * 2 * H_ * 4);
  float* rwF     = (float*)alloc((size_t)M_ * 2 * H_ * 4);
  float* ctxRow  = (float*)alloc(nBSD * 4);
  float* ctxColP = (float*)alloc(nBSD * 4);
  u16*   ctxB    = (u16*)alloc(nBSD * 2);
  int*   permI   = (int*)alloc((size_t)B_ * S_ * 4);
  int*   invpI   = (int*)alloc((size_t)B_ * S_ * 4);
  int*   pcI     = (int*)alloc((size_t)B_ * S_ * 4);
  float* padcF   = (float*)alloc((size_t)B_ * S_ * 4);
  int*   maxvI   = (int*)alloc(B_ * 4);

  auto cast = [&](const float* s, u16* d, long n) {
    int blocks = (int)((n + 255) / 256); if (blocks > 4096) blocks = 4096;
    hipLaunchKernelGGL(cast_f32_to_bf16, dim3(blocks), dim3(256), 0, stream, s, d, n);
  };
  auto packw = [&](const float* s, u32* d, int K, int N) {
    long n = (long)N * (K / 2);
    int blocks = (int)((n + 255) / 256); if (blocks > 4096) blocks = 4096;
    hipLaunchKernelGGL(pack_weight, dim3(blocks), dim3(256), 0, stream, s, d, K, N);
  };
  auto gemm = [&](const u16* A, const u32* Bpk, const float* bias, void* out,
                  int Mm, int Nn, int Kk, int mode) {
    dim3 grid((Nn + BLK_N - 1) / BLK_N, (Mm + BLK_M - 1) / BLK_M);
    hipLaunchKernelGGL(gemm_bf16_wmma, grid, dim3(256), 0, stream, A, Bpk, bias, out, Mm, Nn, Kk, mode);
  };

  // 0) init scalar outputs
  hipLaunchKernelGGL(init_misc, dim3(1), dim3(32), 0, stream, entOut, maxvI);

  // 1) casts (A operands) + weight packing (B operands)
  cast(query,  qbf, nBSD);  cast(keyin, kbf, nBSD);
  cast(value,  vbf, nBSD);  cast(hidden, hbf, nBSD);
  packw(Wq, wqp, D_, D_);   packw(Wk, wkp, D_, D_);
  packw(Wv, wvp, D_, D_);   packw(Wo, wop, D_, D_);
  packw(Wup, wupp, D_, 4 * D_);
  packw(Wgt, wgtp, D_, 4 * D_);
  packw(Wdn, wdnp, 4 * D_, 2 * H_);

  // 2) Q/K/V projections, scattered into [B,H,S,DH]
  gemm(qbf, wqp, bq, Qh, M_, D_, D_, GEMM_OUT_HEADS);
  gemm(kbf, wkp, bk, Kh, M_, D_, D_, GEMM_OUT_HEADS);
  gemm(vbf, wvp, bv, Vh, M_, D_, D_, GEMM_OUT_HEADS);

  // 3) router MLP
  gemm(hbf, wupp, bup, upb, M_, 4 * D_, D_, GEMM_OUT_BF16);
  gemm(hbf, wgtp, bgt, gtb, M_, 4 * D_, D_, GEMM_OUT_BF16);
  {
    long n = (long)M_ * 4 * D_;
    hipLaunchKernelGGL(silu_mul_kernel, dim3(4096), dim3(256), 0, stream, upb, gtb, actb, n);
  }
  gemm(actb, wdnp, bdn, logitsF, M_, 2 * H_, 4 * D_, GEMM_OUT_F32);
  hipLaunchKernelGGL(router_softmax_entropy, dim3((M_ * H_ + 255) / 256), dim3(256), 0, stream,
                     logitsF, rwF, entOut);

  // 4) permutation for column view
  {
    int blocks = (B_ * S_ + 255) / 256;
    hipLaunchKernelGGL(rowmax_kernel, dim3(blocks), dim3(256), 0, stream, posIdx, maxvI);
    hipLaunchKernelGGL(perm_kernel, dim3(blocks), dim3(256), 0, stream, posIdx, maxvI, permI, invpI);
    hipLaunchKernelGGL(pcpad_kernel, dim3(blocks), dim3(256), 0, stream, permI, posIdx, padM, pcI, padcF);
  }
  {
    int blocks = 4096;
    hipLaunchKernelGGL(gather_heads, dim3(blocks), dim3(256), 0, stream, Qh, permI, Qc);
    hipLaunchKernelGGL(gather_heads, dim3(blocks), dim3(256), 0, stream, Kh, permI, Kc);
    hipLaunchKernelGGL(pack_v_bf16, dim3(2048), dim3(256), 0, stream, Vh, (const int*)nullptr, Vph);
    hipLaunchKernelGGL(pack_v_bf16, dim3(2048), dim3(256), 0, stream, Vh, permI, Vpc);
  }

  // 5) attention, both views
  dim3 fgrid(S_ / 16, H_, B_);
  hipLaunchKernelGGL(flash_attn_wmma, fgrid, dim3(32), 0, stream,
                     Qh, Kh, Vph, ctxRow, posBias, (const float*)nullptr, (const int*)nullptr, padM, 0);
  hipLaunchKernelGGL(flash_attn_wmma, fgrid, dim3(32), 0, stream,
                     Qc, Kc, Vpc, ctxColP, (const float*)nullptr, relT, pcI, padcF, 1);

  // 6) combine with routing weights (+ inverse permutation of column context)
  hipLaunchKernelGGL(combine_kernel, dim3(4096), dim3(256), 0, stream,
                     ctxRow, ctxColP, rwF, invpI, ctxB);

  // 7) output projection -> d_out (fp32)
  gemm(ctxB, wop, bo, outp, M_, D_, D_, GEMM_OUT_F32);
}
